// MultiHeadCoupledAttention_56581899157861
// MI455X (gfx1250) — matrix operand order repaired and checked by reading, not compiled, not run on hardware
//
#include <hip/hip_runtime.h>
#include <hip/hip_bf16.h>
#include <cstdint>
#include <cstddef>

// ---------------------------------------------------------------------------
// MultiHeadCoupledAttention for MI455X (gfx1250, wave32, WMMA bf16)
// B=2 L=2048 D_MODEL=1024 H=16 HD=64
// d_out = [ out (B*L*1024 f32) | attn (B*H*L*L f32) ]
// WMMA loops are double-buffered AND pinned with sched_group_barrier so the
// scheduler cannot re-roll the pipeline into load->wait(0)->wmma pairs.
// ---------------------------------------------------------------------------

typedef __attribute__((ext_vector_type(16))) __bf16 v16bf;
typedef __attribute__((ext_vector_type(8)))  __bf16 v8bf;
typedef __attribute__((ext_vector_type(8)))  float  v8f;

union BF16x16 { v16bf v; v8bf h[2]; };

constexpr int SEQ    = 2048;
constexpr int DMODEL = 1024;
constexpr int NH     = 16;
constexpr int HD     = 64;
constexpr int BATCH  = 2;
constexpr int ROWS   = BATCH * SEQ;                       // 4096
constexpr size_t BHLD      = (size_t)BATCH * NH * SEQ * HD;   // 4,194,304
constexpr size_t OUT_ELEMS = (size_t)BATCH * SEQ * DMODEL;    // 4,194,304
constexpr int ATTN_SMEM = 16 * SEQ * 4 + 16 * SEQ * 2;        // 196608 B

// sched_group_barrier masks
#define SGB_WMMA      0x008
#define SGB_VMEM_READ 0x020
#define SGB_DS_READ   0x100
#define SGB_DS_WRITE  0x200

__device__ __forceinline__ __bf16 f2bf(float f) {
  unsigned u = __builtin_bit_cast(unsigned, f);
  u += 0x7FFFu + ((u >> 16) & 1u);                        // round-to-nearest-even
  unsigned short s = (unsigned short)(u >> 16);
  return __builtin_bit_cast(__bf16, s);
}

__device__ __forceinline__ v8f wmma_bf16(v16bf a, v16bf b, v8f c) {
  // D(16x16 f32) = A(16x32 bf16) * B(32x16 bf16) + C
  return __builtin_amdgcn_wmma_f32_16x16x32_bf16(false, a, false, b, (short)0, c,
                                                 false, false);
}

// Double-buffered 16(M)x64(N) x K=1024 GEMM core.
// Arow: this lane's A row (M = lidx for both lane halves).
// Wn0:  weight row block start (row n = Wn0 + (t*16+lidx)*DMODEL).
__device__ __forceinline__ void gemm_16x64(const __bf16* __restrict__ Arow,
                                           const __bf16* __restrict__ Wn0,
                                           int hh, int lidx, v8f (&acc)[4]) {
  BF16x16 a[2];
  BF16x16 bb[2][4];
  auto load = [&](int k, int buf) {
    a[buf].h[0] = *(const v8bf*)(Arow + k + hh * 16);      // A: K=hh*16+{0..7}
    a[buf].h[1] = *(const v8bf*)(Arow + k + hh * 16 + 8);  //    K=hh*16+8..15
#pragma unroll
    for (int t = 0; t < 4; ++t) {                          // B: lane=N, K=hh*16..+15
      const __bf16* Wr = Wn0 + (size_t)(t * 16 + lidx) * DMODEL + k + hh * 16;
      bb[buf][t].h[0] = *(const v8bf*)(Wr);
      bb[buf][t].h[1] = *(const v8bf*)(Wr + 8);
    }
  };
  load(0, 0);
  int buf = 0;
  for (int k = 0; k < DMODEL - 32; k += 32) {
    load(k + 32, buf ^ 1);                                 // next K-step in flight
#pragma unroll
    for (int t = 0; t < 4; ++t) acc[t] = wmma_bf16(a[buf].v, bb[buf][t].v, acc[t]);
    // pin pipeline shape: 10 loads (next iter) first, then 4 WMMAs (this iter)
    __builtin_amdgcn_sched_group_barrier(SGB_VMEM_READ, 10, 0);
    __builtin_amdgcn_sched_group_barrier(SGB_WMMA, 4, 0);
    buf ^= 1;
  }
#pragma unroll
  for (int t = 0; t < 4; ++t) acc[t] = wmma_bf16(a[buf].v, bb[buf][t].v, acc[t]);
}

// ---------------------------- stage 0: converts ----------------------------

__global__ void k_cvt_bf16(const float* __restrict__ src, __bf16* __restrict__ dst,
                           int n) {
  int i = blockIdx.x * blockDim.x + threadIdx.x;
  int stride = gridDim.x * blockDim.x;
  for (; i < n; i += stride) dst[i] = f2bf(src[i]);
}

// vT[b][h][d][s] = query[b][s][h*64+d]  (so attn@V B-operand is K-contiguous)
__global__ void k_build_vT(const float* __restrict__ q, __bf16* __restrict__ vT) {
  int i = blockIdx.x * blockDim.x + threadIdx.x;
  int stride = gridDim.x * blockDim.x;
  for (; i < (int)(BATCH * NH * HD * SEQ); i += stride) {
    int s = i & (SEQ - 1);
    int d = (i >> 11) & (HD - 1);
    int h = (i >> 17) & (NH - 1);
    int b = i >> 21;
    vT[i] = f2bf(q[((size_t)(b * SEQ + s)) * DMODEL + h * HD + d]);
  }
}

// -------------------- stage 1: projection GEMM + RoPE ----------------------
// Wave computes a 16(M) x 64(N) tile = exactly one head => rotate_half partner
// (d +/- 32) is in a sibling accumulator of the SAME lane (pure register op).

__global__ void __launch_bounds__(256)
k_proj_rope(const __bf16* __restrict__ X, const __bf16* __restrict__ W4,
            __bf16* __restrict__ projs) {
  const int lane = threadIdx.x & 31;
  const int wave = threadIdx.x >> 5;
  const int hh   = lane >> 4;        // K-group (lanes 0-15 vs 16-31)
  const int lidx = lane & 15;
  const int head = blockIdx.x;                   // N block of 64 = head
  const int m0   = blockIdx.y * 128 + wave * 16; // row tile
  const int proj = blockIdx.z;

  const __bf16* Arow = X + (size_t)(m0 + lidx) * DMODEL;
  const __bf16* Wn0  = W4 + (size_t)proj * DMODEL * DMODEL + (size_t)head * 64 * DMODEL;

  v8f acc[4] = {};
  gemm_16x64(Arow, Wn0, hh, lidx, acc);

  // RoPE: angle(pos,d) = pos * 10000^{-(d%32)/32}
  float res[4][8];
#pragma unroll
  for (int t = 0; t < 4; ++t) {
    int d = t * 16 + lidx;
    float freq = __expf((float)(d & 31) * (-9.210340371976184f / 32.f));
#pragma unroll
    for (int r = 0; r < 8; ++r) {
      int m = m0 + hh * 8 + r;
      float ang = (float)(m & (SEQ - 1)) * freq;
      float sn, cs;
      __sincosf(ang, &sn, &cs);
      float x  = acc[t][r];
      float xr = (t < 2) ? -acc[t + 2][r] : acc[t - 2][r];
      res[t][r] = x * cs + xr * sn;
    }
  }
#pragma unroll
  for (int t = 0; t < 4; ++t) {
    int d = t * 16 + lidx;
#pragma unroll
    for (int r = 0; r < 8; ++r) {
      int m = m0 + hh * 8 + r;
      int b = m >> 11, pos = m & (SEQ - 1);
      projs[(size_t)proj * BHLD + ((size_t)(b * NH + head) * SEQ + pos) * HD + d] =
          f2bf(res[t][r]);
    }
  }
}

// --------------------------- stage 2: attention ----------------------------
// One workgroup = 16 query rows of one (b,h). Whole 16x2048 score strip lives
// in LDS (128 KB f32) + 16x2048 bf16 prob strip (64 KB) -> 192 KB < 320 KB WGP.

__global__ void __launch_bounds__(256)
k_attention(const __bf16* __restrict__ projs, const __bf16* __restrict__ vT,
            const int* __restrict__ mask, float* __restrict__ attn,
            __bf16* __restrict__ oh) {
  extern __shared__ char smem[];
  float*  S = (float*)smem;                               // [16][2048] scores
  __bf16* P = (__bf16*)(smem + 16 * SEQ * sizeof(float)); // [16][2048] probs

  const int lane = threadIdx.x & 31;
  const int wave = threadIdx.x >> 5;
  const int hh   = lane >> 4;
  const int lidx = lane & 15;
  const int bh   = blockIdx.x;                 // b*16+h
  const int b    = bh >> 4, h = bh & 15;
  const int row0 = blockIdx.y * 16;

  const __bf16* qw = projs + 0 * BHLD + ((size_t)bh * SEQ + row0) * HD;
  const __bf16* kw = projs + 1 * BHLD + ((size_t)bh * SEQ) * HD;
  const __bf16* qp = projs + 2 * BHLD + ((size_t)bh * SEQ + row0) * HD;
  const __bf16* kp = projs + 3 * BHLD + ((size_t)bh * SEQ) * HD;

  // Preload A operands (16x64 each, two 16x32 K-chunks)
  BF16x16 aqw[2], aqp[2];
#pragma unroll
  for (int c = 0; c < 2; ++c) {
    const __bf16* r1 = qw + (size_t)lidx * HD + c * 32 + hh * 16;
    aqw[c].h[0] = *(const v8bf*)(r1);
    aqw[c].h[1] = *(const v8bf*)(r1 + 8);
    const __bf16* r2 = qp + (size_t)lidx * HD + c * 32 + hh * 16;
    aqp[c].h[0] = *(const v8bf*)(r2);
    aqp[c].h[1] = *(const v8bf*)(r2 + 8);
  }

  // scores: silu(qw.kp^T/8) + silu(qp.kw^T/8), * 1/sqrt(2), masked.
  // Wave handles 16 column tiles; key tiles double-buffered across iterations.
  BF16x16 bkp[2][2], bkw[2][2];
  auto loadK = [&](int ct, int buf) {
    int s0 = ct * 16;
#pragma unroll
    for (int c = 0; c < 2; ++c) {
      const __bf16* r1 = kp + (size_t)(s0 + lidx) * HD + c * 32 + hh * 16;
      bkp[buf][c].h[0] = *(const v8bf*)(r1);
      bkp[buf][c].h[1] = *(const v8bf*)(r1 + 8);
      const __bf16* r2 = kw + (size_t)(s0 + lidx) * HD + c * 32 + hh * 16;
      bkw[buf][c].h[0] = *(const v8bf*)(r2);
      bkw[buf][c].h[1] = *(const v8bf*)(r2 + 8);
    }
  };
  auto scoreTile = [&](int ct, int buf) {
    int s0 = ct * 16;
    v8f awp = {}, apw = {};
    awp = wmma_bf16(aqw[0].v, bkp[buf][0].v, awp);
    awp = wmma_bf16(aqw[1].v, bkp[buf][1].v, awp);
    apw = wmma_bf16(aqp[0].v, bkw[buf][0].v, apw);
    apw = wmma_bf16(aqp[1].v, bkw[buf][1].v, apw);
    int mval = mask[b * SEQ + s0 + lidx];
#pragma unroll
    for (int r = 0; r < 8; ++r) {
      float x1 = awp[r] * 0.125f, x2 = apw[r] * 0.125f;
      float sc = (x1 / (1.f + __expf(-x1)) + x2 / (1.f + __expf(-x2))) *
                 0.7071067811865476f;
      if (mval == 0) sc = -INFINITY;
      S[(hh * 8 + r) * SEQ + s0 + lidx] = sc;
    }
  };
  loadK(wave, 0);
  for (int i = 0; i < 15; ++i) {
    loadK(wave + (i + 1) * 8, (i + 1) & 1);   // next tile in flight
    scoreTile(wave + i * 8, i & 1);
    // pipeline: 8 key loads (next), 4 WMMAs (this), 8 LDS stores (this)
    __builtin_amdgcn_sched_group_barrier(SGB_VMEM_READ, 8, 0);
    __builtin_amdgcn_sched_group_barrier(SGB_WMMA, 4, 0);
    __builtin_amdgcn_sched_group_barrier(SGB_DS_WRITE, 8, 0);
  }
  scoreTile(wave + 15 * 8, 15 & 1);
  __syncthreads();

  // single-pass softmax (wave per 2 rows); write attn to HBM exactly once
  float* attn_base = attn + ((size_t)bh * SEQ + row0) * SEQ;
#pragma unroll
  for (int rr = 0; rr < 2; ++rr) {
    int row = wave * 2 + rr;
    float* Sr = S + row * SEQ;
    float mx = -INFINITY;
    for (int c = lane; c < SEQ; c += 32) mx = fmaxf(mx, Sr[c]);
#pragma unroll
    for (int o = 16; o > 0; o >>= 1) mx = fmaxf(mx, __shfl_xor(mx, o, 32));
    float sum = 0.f;
    for (int c = lane; c < SEQ; c += 32) {
      float e = __expf(Sr[c] - mx);
      Sr[c] = e;
      sum += e;
    }
#pragma unroll
    for (int o = 16; o > 0; o >>= 1) sum += __shfl_xor(sum, o, 32);
    float inv = 1.f / sum;
    float* ar = attn_base + (size_t)row * SEQ;
    __bf16* Pr = P + row * SEQ;
    for (int c = lane; c < SEQ; c += 32) {
      float p = Sr[c] * inv;
      ar[c] = p;                      // coalesced 128B stores: the roofline term
      Pr[c] = f2bf(p);
    }
  }
  __syncthreads();

  // reuse first 16x64 of S as f32 accumulator for attn @ V
  for (int i = threadIdx.x; i < 16 * 64; i += 256) S[i] = 0.f;
  __syncthreads();

  // P @ V: A from LDS, B (vT) double-buffered from global; wave owns K=256.
  const __bf16* vTp = vT + (size_t)bh * HD * SEQ;   // [64][2048], K-contiguous
  BF16x16 bv[2][4];
  auto loadV = [&](int ks, int buf) {
    int k0 = wave * 256 + ks * 32;
#pragma unroll
    for (int t = 0; t < 4; ++t) {
      const __bf16* vr = vTp + (size_t)(t * 16 + lidx) * SEQ + k0 + hh * 16;
      bv[buf][t].h[0] = *(const v8bf*)(vr);
      bv[buf][t].h[1] = *(const v8bf*)(vr + 8);
    }
  };
  loadV(0, 0);
  v8f accv[4] = {};
#pragma unroll
  for (int ks = 0; ks < 8; ++ks) {
    if (ks < 7) loadV(ks + 1, (ks + 1) & 1);
    int k0 = wave * 256 + ks * 32;
    BF16x16 pa;
    const __bf16* pr = P + lidx * SEQ + k0 + hh * 16;
    pa.h[0] = *(const v8bf*)(pr);                   // ds_load_b128 from LDS
    pa.h[1] = *(const v8bf*)(pr + 8);
#pragma unroll
    for (int t = 0; t < 4; ++t) accv[t] = wmma_bf16(pa.v, bv[ks & 1][t].v, accv[t]);
    // pipeline: 8 V loads (next), 2 LDS reads (this), 4 WMMAs (this)
    __builtin_amdgcn_sched_group_barrier(SGB_VMEM_READ, 8, 0);
    __builtin_amdgcn_sched_group_barrier(SGB_DS_READ, 2, 0);
    __builtin_amdgcn_sched_group_barrier(SGB_WMMA, 4, 0);
  }
#pragma unroll
  for (int t = 0; t < 4; ++t)
#pragma unroll
    for (int r = 0; r < 8; ++r)
      atomicAdd(&S[(hh * 8 + r) * 64 + t * 16 + lidx], accv[t][r]); // ds_add_f32
  __syncthreads();

  for (int i = threadIdx.x; i < 16 * 64; i += 256) {
    int m = i >> 6, d = i & 63;
    oh[((size_t)(b * SEQ + row0 + m)) * DMODEL + h * HD + d] = f2bf(S[i]);
  }
}

// ----------------------- stage 3: output projection ------------------------

__global__ void __launch_bounds__(256)
k_out_gemm(const __bf16* __restrict__ X, const __bf16* __restrict__ W,
           float* __restrict__ out) {
  const int lane = threadIdx.x & 31;
  const int wave = threadIdx.x >> 5;
  const int hh   = lane >> 4;
  const int lidx = lane & 15;
  const int nb   = blockIdx.x;
  const int m0   = blockIdx.y * 128 + wave * 16;

  const __bf16* Arow = X + (size_t)(m0 + lidx) * DMODEL;
  const __bf16* Wn0  = W + (size_t)nb * 64 * DMODEL;

  v8f acc[4] = {};
  gemm_16x64(Arow, Wn0, hh, lidx, acc);

#pragma unroll
  for (int t = 0; t < 4; ++t) {
    int n = nb * 64 + t * 16 + lidx;
#pragma unroll
    for (int r = 0; r < 8; ++r) {
      int m = m0 + hh * 8 + r;
      out[(size_t)m * DMODEL + n] = acc[t][r];
    }
  }
}

// --------------------------------- launch ----------------------------------

extern "C" void kernel_launch(void* const* d_in, const int* in_sizes, int n_in,
                              void* d_out, int out_size, void* d_ws, size_t ws_size,
                              hipStream_t stream) {
  (void)in_sizes; (void)n_in; (void)out_size; (void)ws_size;

  const float* query = (const float*)d_in[0];
  const int*   mask  = (const int*)d_in[1];
  const float* Wqw   = (const float*)d_in[2];
  const float* Wkw   = (const float*)d_in[3];
  const float* Wqp   = (const float*)d_in[4];
  const float* Wkp   = (const float*)d_in[5];
  const float* Wout  = (const float*)d_in[6];

  char* ws = (char*)d_ws;                         // ~66 MB total
  __bf16* q_bf   = (__bf16*)(ws);                 //  8 MB  query bf16
  __bf16* w_bf   = (__bf16*)(ws + (8u  << 20));   //  8 MB  4 proj weights bf16
  __bf16* projs  = (__bf16*)(ws + (16u << 20));   // 32 MB  qw/kw/qp/kp roped
  __bf16* vT     = (__bf16*)(ws + (48u << 20));   //  8 MB  V transposed
  __bf16* wout_b = (__bf16*)(ws + (56u << 20));   //  2 MB  Wout bf16
  __bf16* oh_bf  = (__bf16*)(ws + (58u << 20));   //  8 MB  attn@V result bf16

  float* out_f   = (float*)d_out;
  float* attn_f  = out_f + OUT_ELEMS;

  // 192 KB dynamic LDS for the attention kernel
  hipFuncSetAttribute(reinterpret_cast<const void*>(k_attention),
                      hipFuncAttributeMaxDynamicSharedMemorySize, ATTN_SMEM);

  k_cvt_bf16<<<1024, 256, 0, stream>>>(query, q_bf, ROWS * DMODEL);
  k_cvt_bf16<<<512, 256, 0, stream>>>(Wqw, w_bf + 0 * (size_t)DMODEL * DMODEL, DMODEL * DMODEL);
  k_cvt_bf16<<<512, 256, 0, stream>>>(Wkw, w_bf + 1 * (size_t)DMODEL * DMODEL, DMODEL * DMODEL);
  k_cvt_bf16<<<512, 256, 0, stream>>>(Wqp, w_bf + 2 * (size_t)DMODEL * DMODEL, DMODEL * DMODEL);
  k_cvt_bf16<<<512, 256, 0, stream>>>(Wkp, w_bf + 3 * (size_t)DMODEL * DMODEL, DMODEL * DMODEL);
  k_cvt_bf16<<<512, 256, 0, stream>>>(Wout, wout_b, DMODEL * DMODEL);
  k_build_vT<<<1024, 256, 0, stream>>>(query, vT);

  k_proj_rope<<<dim3(16, 32, 4), 256, 0, stream>>>(q_bf, w_bf, projs);
  k_attention<<<dim3(32, 128), 256, ATTN_SMEM, stream>>>(projs, vT, mask, attn_f, oh_bf);
  k_out_gemm<<<dim3(16, 32), 256, 0, stream>>>(oh_bf, wout_b, out_f);
}
